// MultiHeadAttention_9835475108183
// MI455X (gfx1250) — matrix operand order repaired and checked by reading, not compiled, not run on hardware
//
#include <hip/hip_runtime.h>
#include <hip/hip_bf16.h>

// ---------------------------------------------------------------------------
// MultiHeadAttention for MI455X (gfx1250, wave32, WMMA).
// B=4, S=2048, D=1024, H=16, DK=DV=64.
// All GEMM stages run on v_wmma_f32_16x16x32_bf16 (fp32 accumulate).
// Operand layouts keep per-lane data contiguous (ds_load_b128):
//   A operand: row-major [M][K]; B operand: n-major [N][K].
// Weights pre-transposed to [N][K]; V produced transposed ([v][t]).
// All GEMM kernels double-buffer their LDS tiles: one barrier per K-step,
// global->LDS loads for step i+1 overlap the WMMA chain of step i.
// ---------------------------------------------------------------------------

typedef __bf16 bf16_t;
typedef bf16_t bf16x16 __attribute__((ext_vector_type(16)));
typedef bf16_t bf16x8  __attribute__((ext_vector_type(8)));
typedef float  floatx8 __attribute__((ext_vector_type(8)));

#define BB   4
#define SS   2048
#define DDIM 1024
#define HH   16
#define DKV  64

#define WMMA_BF16(a, b, c) \
  __builtin_amdgcn_wmma_f32_16x16x32_bf16(false, (a), false, (b), (short)0, (c), false, false)

__device__ __forceinline__ bf16x16 cat8(bf16x8 lo, bf16x8 hi) {
  return __builtin_shufflevector(lo, hi, 0, 1, 2, 3, 4, 5, 6, 7,
                                 8, 9, 10, 11, 12, 13, 14, 15);
}

// A operand (16x32 bf16): lane holds row m = lane&15; its 16 values are the
// contiguous K-range [16*half+koff, +16): one 32B run (matches B).
__device__ __forceinline__ bf16x16 loadA_vec(const bf16_t* base, int ld,
                                             int lane, int koff) {
  const int hgrp = lane >> 4, m = lane & 15;
  const bf16x8* p = (const bf16x8*)(base + (size_t)m * ld + 16 * hgrp + koff);
  return cat8(p[0], p[1]);  // one contiguous run
}

// B operand (32x16 bf16): lane holds column n = lane&15; its 16 values are the
// contiguous K-range [16*half+koff, +16): one 32B run. Source is [N][K].
__device__ __forceinline__ bf16x16 loadB_vec(const bf16_t* base, int ld,
                                             int lane, int koff) {
  const int hgrp = lane >> 4, n = lane & 15;
  const bf16x8* p = (const bf16x8*)(base + (size_t)n * ld + 16 * hgrp + koff);
  return cat8(p[0], p[1]);
}

// --- Kernel 1a: fp32 -> bf16 (straight) -------------------------------------

__global__ void f32_to_bf16_kernel(const float* __restrict__ src,
                                   bf16_t* __restrict__ dst, int nelem) {
  int i      = blockIdx.x * blockDim.x + threadIdx.x;
  int stride = gridDim.x * blockDim.x;
  for (; i < nelem; i += stride) dst[i] = (bf16_t)src[i];
}

// --- Kernel 1b: fp32 -> bf16 with [b][r][c] -> [b][c][r] transpose ----------

__global__ void f32_to_bf16_transpose_kernel(const float* __restrict__ src,
                                             bf16_t* __restrict__ dst,
                                             int R, int C, int batch) {
  int total  = batch * R * C;
  int stride = gridDim.x * blockDim.x;
  for (int i = blockIdx.x * blockDim.x + threadIdx.x; i < total; i += stride) {
    int b  = i / (R * C);
    int rc = i - b * (R * C);
    int c  = rc / R;
    int r  = rc - c * R;
    dst[i] = (bf16_t)src[(size_t)b * R * C + (size_t)r * C + c];
  }
}

// --- Kernel 2: QKV projection ----------------------------------------------
// Grid: (S/64, B*H). Block: 128 (4 waves). Each wave: 16 rows x 64 cols of
// Q, K, V. Weights arrive pre-transposed [H][64(n)][1024(k)].
// Q, K stored row-major [s][d]; V stored transposed [v][t] (packed b128).

__device__ __forceinline__ void qkv_copy_tiles(
    bf16_t* Xd, bf16_t* W0, bf16_t* W1, bf16_t* W2,
    const bf16_t* xrow, const bf16_t* w0, const bf16_t* w1, const bf16_t* w2,
    int d0, int tid) {
  uint4* xd = (uint4*)Xd;
#pragma unroll
  for (int rep = 0; rep < 2; ++rep) {
    int i = tid + rep * 128;  // 64 rows x 4 uint4
    int row = i >> 2, part = i & 3;
    xd[i] = ((const uint4*)(xrow + (size_t)row * DDIM + d0))[part];
  }
  bf16_t* wdst[3]       = {W0, W1, W2};
  const bf16_t* wsrc[3] = {w0, w1, w2};
#pragma unroll
  for (int mtx = 0; mtx < 3; ++mtx) {
    uint4* wd = (uint4*)wdst[mtx];
    const bf16_t* ws = wsrc[mtx] + d0;  // row n, stride DDIM
#pragma unroll
    for (int rep = 0; rep < 2; ++rep) {
      int i = tid + rep * 128;
      int row = i >> 2, part = i & 3;
      wd[i] = ((const uint4*)(ws + (size_t)row * DDIM))[part];
    }
  }
}

__global__ __launch_bounds__(128) void qkv_proj_kernel(
    const bf16_t* __restrict__ xb,
    const bf16_t* __restrict__ wqT, const bf16_t* __restrict__ wkT,
    const bf16_t* __restrict__ wvT,
    bf16_t* __restrict__ qb, bf16_t* __restrict__ kb,
    bf16_t* __restrict__ vbT) {
  __shared__ __align__(16) bf16_t Xs[2][64 * 32];     // [m][k]
  __shared__ __align__(16) bf16_t Ws[2][3][64 * 32];  // [n][k]

  const int tid = threadIdx.x, lane = tid & 31, wid = tid >> 5;
  const int s0 = blockIdx.x * 64;
  const int bh = blockIdx.y;
  const int h  = bh % HH;
  const int b  = bh / HH;

  const bf16_t* xrow = xb + ((size_t)b * SS + s0) * DDIM;
  const bf16_t* w0 = wqT + (size_t)h * DKV * DDIM;
  const bf16_t* w1 = wkT + (size_t)h * DKV * DDIM;
  const bf16_t* w2 = wvT + (size_t)h * DKV * DDIM;

  floatx8 acc[3][4] = {};

  qkv_copy_tiles(Xs[0], Ws[0][0], Ws[0][1], Ws[0][2],
                 xrow, w0, w1, w2, 0, tid);
  __syncthreads();

  for (int it = 0; it < DDIM / 32; ++it) {
    const int cur = it & 1;
    if (it + 1 < DDIM / 32)  // overlap next tile load with this WMMA chain
      qkv_copy_tiles(Xs[cur ^ 1], Ws[cur ^ 1][0], Ws[cur ^ 1][1],
                     Ws[cur ^ 1][2], xrow, w0, w1, w2, (it + 1) * 32, tid);

    bf16x16 a = loadA_vec(Xs[cur] + wid * 16 * 32, 32, lane, 0);
#pragma unroll
    for (int mtx = 0; mtx < 3; ++mtx) {
      bf16x16 bm[4];
#pragma unroll
      for (int nt = 0; nt < 4; ++nt)
        bm[nt] = loadB_vec(Ws[cur][mtx] + nt * 16 * 32, 32, lane, 0);
#pragma unroll
      for (int nt = 0; nt < 4; ++nt)
        acc[mtx][nt] = WMMA_BF16(a, bm[nt], acc[mtx][nt]);
    }
    __syncthreads();
  }

  const int hgrp = lane >> 4, n = lane & 15;

  // Q, K: row-major [s][d] (A / B(n-major over t) in flash).
  bf16_t* qkout[2] = {qb, kb};
#pragma unroll
  for (int mtx = 0; mtx < 2; ++mtx) {
    bf16_t* base = qkout[mtx] + ((size_t)bh * SS + s0 + wid * 16) * DKV;
#pragma unroll
    for (int nt = 0; nt < 4; ++nt)
#pragma unroll
      for (int r = 0; r < 8; ++r)
        base[(size_t)(r + 8 * hgrp) * DKV + nt * 16 + n] =
            (bf16_t)acc[mtx][nt][r];
  }
  // V transposed [bh][v][t]: the 8 accumulator rows per lane are consecutive
  // t -> one packed 16B store per n-tile.
#pragma unroll
  for (int nt = 0; nt < 4; ++nt) {
    bf16x8 pk;
#pragma unroll
    for (int r = 0; r < 8; ++r) pk[r] = (bf16_t)acc[2][nt][r];
    bf16_t* addr = vbT + ((size_t)bh * DKV + nt * 16 + n) * SS +
                   (s0 + wid * 16 + 8 * hgrp);
    *(bf16x8*)addr = pk;
  }
}

// --- Kernel 3: flash attention ---------------------------------------------
// Grid: (S/64, B*H). Block: 128 (4 waves); each wave owns 16 query rows.
// Double-buffered K/V tiles: one barrier per key tile. P bounces through a
// wave-private LDS tile (LDS is in-order per wave -> no block barrier needed).

__device__ __forceinline__ void attn_copy_tiles(
    bf16_t* Kd, bf16_t* Vd, const bf16_t* kbh, const bf16_t* vbh,
    int t0, int tid) {
  uint4* kd = (uint4*)Kd;
  uint4* vd = (uint4*)Vd;
  const uint4* ks = (const uint4*)(kbh + (size_t)t0 * DKV);  // 8KB slab
#pragma unroll
  for (int rep = 0; rep < 4; ++rep) {
    int i = tid + rep * 128;
    kd[i] = ks[i];
    int vrow = i >> 3, part = i & 7;  // 64 rows x 8 uint4, row stride SS
    vd[i] = ((const uint4*)(vbh + (size_t)vrow * SS + t0))[part];
  }
}

__global__ __launch_bounds__(128) void flash_attn_kernel(
    const bf16_t* __restrict__ qb, const bf16_t* __restrict__ kb,
    const bf16_t* __restrict__ vbT, bf16_t* __restrict__ ob,
    const int* __restrict__ has_mask_p) {
  __shared__ __align__(16) bf16_t Kt[2][64 * 64];   // [t][d] == [n][k]
  __shared__ __align__(16) bf16_t Vt[2][64 * 64];   // [v][t] == [n][k]
  __shared__ __align__(16) bf16_t Ps[4][16 * 64];   // per-wave [m][t]

  const int tid = threadIdx.x, lane = tid & 31, wid = tid >> 5;
  const int qt = blockIdx.x;
  const int bh = blockIdx.y;
  const int h  = bh % HH;
  const int b  = bh / HH;
  const int hm = has_mask_p[0];
  const int nkt = hm ? (qt + 1) : (SS / 64);

  const bf16_t* kbh = kb + (size_t)bh * SS * DKV;
  const bf16_t* vbh = vbT + (size_t)bh * DKV * SS;

  const bf16_t* qbase = qb + ((size_t)bh * SS + qt * 64 + wid * 16) * DKV;
  bf16x16 qa0 = loadA_vec(qbase, DKV, lane, 0);
  bf16x16 qa1 = loadA_vec(qbase, DKV, lane, 32);

  floatx8 o[4] = {};
  float mrow[8], lrow[8];
#pragma unroll
  for (int r = 0; r < 8; ++r) { mrow[r] = -3.0e38f; lrow[r] = 0.0f; }

  const int hgrp = lane >> 4, n = lane & 15;

  attn_copy_tiles(Kt[0], Vt[0], kbh, vbh, 0, tid);
  __syncthreads();

  for (int kt = 0; kt < nkt; ++kt) {
    const int cur = kt & 1;
    if (kt + 1 < nkt)  // overlap next tile load with this tile's compute
      attn_copy_tiles(Kt[cur ^ 1], Vt[cur ^ 1], kbh, vbh, (kt + 1) * 64, tid);
    if (kt + 2 < nkt) {  // global_prefetch one tile further out
      __builtin_prefetch((const char*)(kbh + (size_t)(kt + 2) * 64 * DKV) +
                             tid * 64, 0, 1);
      __builtin_prefetch((const char*)(vbh + (size_t)(tid >> 1) * SS +
                                       (kt + 2) * 64 + (tid & 1) * 32), 0, 1);
    }

    // scores = Q * K^T
    floatx8 s[4] = {};
    {
      bf16x16 bk[4];
#pragma unroll
      for (int nt = 0; nt < 4; ++nt)
        bk[nt] = loadB_vec(Kt[cur] + nt * 16 * DKV, DKV, lane, 0);
#pragma unroll
      for (int nt = 0; nt < 4; ++nt) s[nt] = WMMA_BF16(qa0, bk[nt], s[nt]);
#pragma unroll
      for (int nt = 0; nt < 4; ++nt)
        bk[nt] = loadB_vec(Kt[cur] + nt * 16 * DKV, DKV, lane, 32);
#pragma unroll
      for (int nt = 0; nt < 4; ++nt) s[nt] = WMMA_BF16(qa1, bk[nt], s[nt]);
    }

    const float scale = 0.125f;  // 1/sqrt(64)
    const bool diag = (hm != 0) && (kt == qt);
#pragma unroll
    for (int nt = 0; nt < 4; ++nt) {
#pragma unroll
      for (int r = 0; r < 8; ++r) {
        float sv = s[nt][r] * scale;
        if (diag) {
          int col = nt * 16 + n;
          int row = wid * 16 + r + 8 * hgrp;
          sv = (col > row) ? -1.0e30f : sv;
        }
        s[nt][r] = sv;
      }
    }

    // Online softmax: row M lives in vgpr r across the 16 lanes of a
    // half-wave -> shfl_xor masks 1,2,4,8 reduce within a row.
    float alpha[8], mnew[8];
#pragma unroll
    for (int r = 0; r < 8; ++r) {
      float vmx = fmaxf(fmaxf(s[0][r], s[1][r]), fmaxf(s[2][r], s[3][r]));
      vmx = fmaxf(vmx, __shfl_xor(vmx, 1));
      vmx = fmaxf(vmx, __shfl_xor(vmx, 2));
      vmx = fmaxf(vmx, __shfl_xor(vmx, 4));
      vmx = fmaxf(vmx, __shfl_xor(vmx, 8));
      mnew[r]  = fmaxf(mrow[r], vmx);
      alpha[r] = __expf(mrow[r] - mnew[r]);
      mrow[r]  = mnew[r];
    }

    float rsum[8];
#pragma unroll
    for (int r = 0; r < 8; ++r) rsum[r] = 0.0f;
#pragma unroll
    for (int nt = 0; nt < 4; ++nt) {
#pragma unroll
      for (int r = 0; r < 8; ++r) {
        float p = __expf(s[nt][r] - mnew[r]);
        rsum[r] += p;
        Ps[wid][(r + 8 * hgrp) * 64 + nt * 16 + n] = (bf16_t)p;
      }
    }
#pragma unroll
    for (int r = 0; r < 8; ++r) {
      float t = rsum[r];
      t += __shfl_xor(t, 1);
      t += __shfl_xor(t, 2);
      t += __shfl_xor(t, 4);
      t += __shfl_xor(t, 8);
      lrow[r] = lrow[r] * alpha[r] + t;
    }
#pragma unroll
    for (int nt = 0; nt < 4; ++nt)
#pragma unroll
      for (int r = 0; r < 8; ++r) o[nt][r] *= alpha[r];

    // O += P * V  (Ps is wave-private; LDS in-order per wave, no barrier)
#pragma unroll
    for (int kc = 0; kc < 2; ++kc) {
      bf16x16 pa = loadA_vec(Ps[wid], 64, lane, kc * 32);
      bf16x16 bv[4];
#pragma unroll
      for (int nt = 0; nt < 4; ++nt)
        bv[nt] = loadB_vec(Vt[cur] + nt * 16 * 64, 64, lane, kc * 32);
#pragma unroll
      for (int nt = 0; nt < 4; ++nt) o[nt] = WMMA_BF16(pa, bv[nt], o[nt]);
    }
    __syncthreads();  // buffer swap: everyone done with tile[cur]
  }

  float inv[8];
#pragma unroll
  for (int r = 0; r < 8; ++r) inv[r] = (lrow[r] > 0.0f) ? (1.0f / lrow[r]) : 0.0f;

  // Heads-concatenated [b][s][h*DV + v] so the out-proj A reads row-major.
  bf16_t* obase = ob + ((size_t)b * SS + qt * 64 + wid * 16) * (HH * DKV) +
                  h * DKV;
#pragma unroll
  for (int nt = 0; nt < 4; ++nt)
#pragma unroll
    for (int r = 0; r < 8; ++r)
      obase[(size_t)(r + 8 * hgrp) * (HH * DKV) + nt * 16 + n] =
          (bf16_t)(o[nt][r] * inv[r]);
}

// --- Kernel 4: output projection [B*S,1024] x woT[1024(n)][1024(k)] ---------

__device__ __forceinline__ void oproj_copy_tiles(
    bf16_t* Ad, bf16_t* Bd, const bf16_t* asrc, const bf16_t* bsrc,
    int d0, int tid) {
  uint4* ad = (uint4*)Ad;
  uint4* bd = (uint4*)Bd;
#pragma unroll
  for (int rep = 0; rep < 2; ++rep) {
    int i = tid + rep * 128;  // 64 rows x 4 uint4 each
    int row = i >> 2, part = i & 3;
    ad[i] = ((const uint4*)(asrc + (size_t)row * DDIM + d0))[part];
    bd[i] = ((const uint4*)(bsrc + (size_t)row * DDIM + d0))[part];
  }
}

__global__ __launch_bounds__(128) void out_proj_kernel(
    const bf16_t* __restrict__ ob, const bf16_t* __restrict__ woT,
    float* __restrict__ out) {
  __shared__ __align__(16) bf16_t As[2][64 * 32];  // [m][k]
  __shared__ __align__(16) bf16_t Bs[2][64 * 32];  // [n][k]

  const int tid = threadIdx.x, lane = tid & 31, wid = tid >> 5;
  const int n0  = blockIdx.x * 64;
  const int ms0 = blockIdx.y * 64;

  const bf16_t* asrc = ob + (size_t)ms0 * DDIM;
  const bf16_t* bsrc = woT + (size_t)n0 * DDIM;

  floatx8 acc[4] = {};

  oproj_copy_tiles(As[0], Bs[0], asrc, bsrc, 0, tid);
  __syncthreads();

  for (int it = 0; it < DDIM / 32; ++it) {
    const int cur = it & 1;
    if (it + 1 < DDIM / 32)
      oproj_copy_tiles(As[cur ^ 1], Bs[cur ^ 1], asrc, bsrc, (it + 1) * 32,
                       tid);

    bf16x16 a = loadA_vec(As[cur] + wid * 16 * 32, 32, lane, 0);
    bf16x16 bm[4];
#pragma unroll
    for (int nt = 0; nt < 4; ++nt)
      bm[nt] = loadB_vec(Bs[cur] + nt * 16 * 32, 32, lane, 0);
#pragma unroll
    for (int nt = 0; nt < 4; ++nt) acc[nt] = WMMA_BF16(a, bm[nt], acc[nt]);
    __syncthreads();
  }

  const int hgrp = lane >> 4, n = lane & 15;
  float* base = out + (size_t)(ms0 + wid * 16) * DDIM + n0;
#pragma unroll
  for (int nt = 0; nt < 4; ++nt)
#pragma unroll
    for (int r = 0; r < 8; ++r)
      base[(size_t)(r + 8 * hgrp) * DDIM + nt * 16 + n] = acc[nt][r];
}

// --- Host launcher ----------------------------------------------------------

extern "C" void kernel_launch(void* const* d_in, const int* in_sizes, int n_in,
                              void* d_out, int out_size, void* d_ws, size_t ws_size,
                              hipStream_t stream) {
  const float* x  = (const float*)d_in[0];
  const float* wq = (const float*)d_in[1];
  const float* wk = (const float*)d_in[2];
  const float* wv = (const float*)d_in[3];
  const float* wo = (const float*)d_in[4];
  const int* has_mask = (const int*)d_in[5];
  float* out = (float*)d_out;

  // Bump-allocate bf16 scratch (~88 MB) out of d_ws, 256B-aligned.
  size_t off = 0;
  char* wsb = (char*)d_ws;
  auto alloc_bf = [&](size_t nelem) -> bf16_t* {
    bf16_t* p = (bf16_t*)(wsb + off);
    off += ((nelem * sizeof(bf16_t) + 255) & ~(size_t)255);
    return p;
  };
  bf16_t* xb   = alloc_bf((size_t)BB * SS * DDIM);
  bf16_t* wqTb = alloc_bf((size_t)HH * DKV * DDIM);   // [H][n][k]
  bf16_t* wkTb = alloc_bf((size_t)HH * DKV * DDIM);
  bf16_t* wvTb = alloc_bf((size_t)HH * DKV * DDIM);
  bf16_t* woTb = alloc_bf((size_t)DDIM * DDIM);       // [n][k]
  bf16_t* qb   = alloc_bf((size_t)BB * HH * SS * DKV);
  bf16_t* kb   = alloc_bf((size_t)BB * HH * SS * DKV);
  bf16_t* vbT  = alloc_bf((size_t)BB * HH * DKV * SS);  // [bh][v][t]
  bf16_t* obf  = alloc_bf((size_t)BB * SS * HH * DKV);
  (void)ws_size; (void)in_sizes; (void)n_in; (void)out_size;

  f32_to_bf16_kernel<<<2048, 256, 0, stream>>>(x, xb, BB * SS * DDIM);
  f32_to_bf16_transpose_kernel<<<512, 256, 0, stream>>>(wq, wqTb, DDIM, DKV, HH);
  f32_to_bf16_transpose_kernel<<<512, 256, 0, stream>>>(wk, wkTb, DDIM, DKV, HH);
  f32_to_bf16_transpose_kernel<<<512, 256, 0, stream>>>(wv, wvTb, DDIM, DKV, HH);
  f32_to_bf16_transpose_kernel<<<512, 256, 0, stream>>>(wo, woTb, DDIM, DDIM, 1);

  qkv_proj_kernel<<<dim3(SS / 64, BB * HH), 128, 0, stream>>>(
      xb, wqTb, wkTb, wvTb, qb, kb, vbT);
  flash_attn_kernel<<<dim3(SS / 64, BB * HH), 128, 0, stream>>>(
      qb, kb, vbT, obf, has_mask);
  out_proj_kernel<<<dim3(DDIM / 64, (BB * SS) / 64), 128, 0, stream>>>(
      obf, woTb, out);
}
